// Attention_47837345742992
// MI455X (gfx1250) — matrix operand order repaired and checked by reading, not compiled, not run on hardware
//
#include <hip/hip_runtime.h>
#include <hip/hip_bf16.h>
#include <cstdint>

typedef __attribute__((ext_vector_type(16))) _Float16 v16h;
typedef __attribute__((ext_vector_type(8)))  _Float16 v8h;
typedef __attribute__((ext_vector_type(4)))  _Float16 v4h;
typedef __attribute__((ext_vector_type(8)))  float    v8f;
typedef __attribute__((ext_vector_type(4)))  unsigned int v4u;
typedef __attribute__((ext_vector_type(8)))  int      v8i;
typedef __attribute__((ext_vector_type(4)))  int      v4i;

#define WMMA_F16(A, B, C) \
  __builtin_amdgcn_wmma_f32_16x16x32_f16(false, (A), false, (B), (short)0, (C), false, false)

// ---------------------------------------------------------------------------
// TDM: issue a tensor_load_to_lds for a 2-D f16 tile (tile1 rows x tile0 elems,
// row stride stride0 elems). tile1==0 -> 1-D tile. Descriptor packing per
// CDNA5 ISA (D# groups 0/1; groups 2/3 zero => <=2D).
// This toolchain's builtin takes 6 args (g0, g1, g2, g3, g4-spare, cpol).
// ---------------------------------------------------------------------------
__device__ __forceinline__ void tdm_load_f16(unsigned lds_byte_addr,
                                             const void* gaddr,
                                             unsigned tile0, unsigned tile1,
                                             unsigned long long tensor0,
                                             unsigned tensor1,
                                             unsigned long long stride0) {
  unsigned long long ga = (unsigned long long)(uintptr_t)gaddr;
  v4u g0;
  g0.x = 1u;                                              // count=1, user mode
  g0.y = lds_byte_addr;                                   // lds_addr [63:32]
  g0.z = (unsigned)(ga & 0xffffffffull);                  // global_addr[31:0]
  g0.w = (unsigned)((ga >> 32) & 0x1ffffffull) | (2u << 30); // addr[56:32]|type=2
  v8i g1;
  g1[0] = (int)(1u << 16);                                // data_size=1 (2 bytes)
  g1[1] = (int)((tensor0 & 0xffffull) << 16);             // tensor_dim0[15:0]
  g1[2] = (int)(((tensor0 >> 16) & 0xffffull) | ((unsigned long long)(tensor1 & 0xffffu) << 16));
  g1[3] = (int)(((tensor1 >> 16) & 0xffffu) | (tile0 << 16)); // tile_dim0
  g1[4] = (int)(tile1 & 0xffffu);                         // tile_dim1 (tile_dim2=0)
  g1[5] = (int)(stride0 & 0xffffffffull);                 // tensor_dim0_stride[31:0]
  g1[6] = (int)((stride0 >> 32) & 0xffffull);             // stride0[47:32], stride1=0
  g1[7] = 0;
  v4i z4 = {0, 0, 0, 0};
  v8i z8 = {0, 0, 0, 0, 0, 0, 0, 0};
  __builtin_amdgcn_tensor_load_to_lds(g0, g1, z4, z4, z8, 0);
}

// ---------------------------------------------------------------------------
// Kernel 1: projection GEMM  Y = X @ W^T + b   (X:[8192,1024] W:[1024,1024])
// mode 0/1: f16 [a*16+h][seq][64]; mode 2 (V): transposed f16 [a*16+h][64][seq].
// Block: 256 thr (8 waves), tile 64(M) x 128(N), wave tile 32x32, k-step 64.
// ---------------------------------------------------------------------------
__global__ __launch_bounds__(256) void proj_kernel(
    const float* __restrict__ q, const float* __restrict__ k, const float* __restrict__ v,
    const float* __restrict__ Wq, const float* __restrict__ bq,
    const float* __restrict__ Wk, const float* __restrict__ bk,
    const float* __restrict__ Wv, const float* __restrict__ bv,
    _Float16* __restrict__ qp, _Float16* __restrict__ kp, _Float16* __restrict__ vpt)
{
  const int mode = blockIdx.z;
  const float* X  = (mode == 0) ? q  : (mode == 1) ? k  : v;
  const float* W  = (mode == 0) ? Wq : (mode == 1) ? Wk : Wv;
  const float* Bv = (mode == 0) ? bq : (mode == 1) ? bk : bv;
  _Float16* out   = (mode == 0) ? qp : (mode == 1) ? kp : vpt;

  __shared__ __align__(32) _Float16 xs[64 * 64];    // X tile (f16)
  __shared__ __align__(32) _Float16 wsm[128 * 64];  // W tile (f16)

  const int tid  = threadIdx.x;
  const int lane = tid & 31;
  const int wid  = tid >> 5;
  const int l16  = lane & 15;
  const int hsel = (lane < 16) ? 0 : 1;
  const int mloc = (wid >> 2) * 32;               // 0 | 32
  const int nloc = (wid & 3) * 32;                // 0..96
  const int mbase = blockIdx.y * 64;
  const int nbase = blockIdx.x * 128;

  v8f acc[2][2] = {};

  for (int e0 = 0; e0 < 1024; e0 += 64) {
    // Stage X tile 64x64 (fp32 -> f16)
    #pragma unroll
    for (int i = 0; i < 4; ++i) {
      int f = tid + i * 256;                      // 0..1023
      int row = f >> 4, c = (f & 15) * 4;
      float4 xv = *(const float4*)(X + (size_t)(mbase + row) * 1024 + e0 + c);
      v4h hx; hx.x = (_Float16)xv.x; hx.y = (_Float16)xv.y;
      hx.z = (_Float16)xv.z; hx.w = (_Float16)xv.w;
      *(v4h*)&xs[row * 64 + c] = hx;
    }
    // Stage W tile 128x64 (fp32 -> f16)
    #pragma unroll
    for (int i = 0; i < 8; ++i) {
      int f = tid + i * 256;                      // 0..2047
      int row = f >> 4, c = (f & 15) * 4;
      float4 wv = *(const float4*)(W + (size_t)(nbase + row) * 1024 + e0 + c);
      v4h hw; hw.x = (_Float16)wv.x; hw.y = (_Float16)wv.y;
      hw.z = (_Float16)wv.z; hw.w = (_Float16)wv.w;
      *(v4h*)&wsm[row * 64 + c] = hw;
    }
    __syncthreads();

    #pragma unroll
    for (int ks2 = 0; ks2 < 2; ++ks2) {           // two K=32 sub-steps
      v16h afr[2], bfr[2];
      #pragma unroll
      for (int i = 0; i < 2; ++i) {               // A frags (contiguous k = 16*hsel + s)
        int row = mloc + i * 16 + l16;
        int o = ks2 * 32 + hsel * 16;
        *(v8h*)&afr[i]       = *(const v8h*)&xs[row * 64 + o];
        *((v8h*)&afr[i] + 1) = *(const v8h*)&xs[row * 64 + o + 8];
      }
      #pragma unroll
      for (int j = 0; j < 2; ++j) {               // B frags: B[k,n] = W[n,e0+k]
        int row = nloc + j * 16 + l16;
        bfr[j] = *(const v16h*)&wsm[row * 64 + ks2 * 32 + hsel * 16];
      }
      #pragma unroll
      for (int i = 0; i < 2; ++i)
        #pragma unroll
        for (int j = 0; j < 2; ++j)
          acc[i][j] = WMMA_F16(afr[i], bfr[j], acc[i][j]);
    }
    __syncthreads();
  }

  // Epilogue: bias add, f16 convert, scatter to head-major layouts
  #pragma unroll
  for (int j = 0; j < 2; ++j) {
    int nglob = nbase + nloc + j * 16 + l16;
    float bias = Bv[nglob];
    int head = nglob >> 6, c = nglob & 63;
    #pragma unroll
    for (int i = 0; i < 2; ++i) {
      #pragma unroll
      for (int r = 0; r < 8; ++r) {
        int mglob = mbase + mloc + i * 16 + r + hsel * 8;
        int aidx = mglob >> 11, mrow = mglob & 2047;
        _Float16 val = (_Float16)(acc[i][j][r] + bias);
        if (mode == 2)
          out[((size_t)(aidx * 16 + head) * 64 + c) * 2048 + mrow] = val;
        else
          out[((size_t)(aidx * 16 + head) * 2048 + mrow) * 64 + c] = val;
      }
    }
  }
}

// ---------------------------------------------------------------------------
// Kernel 2: flash attention per head; 128 Q-rows per block, 8 waves x 16 rows.
// K/V^T chunks (64 t-cols) DMA'd by the Tensor Data Mover into double-buffered
// LDS; DMA of chunk i+1 overlaps the 16 WMMAs of chunk i. Unscaled QK^T.
// ---------------------------------------------------------------------------
__global__ __launch_bounds__(256) void attn_kernel(
    const _Float16* __restrict__ qp, const _Float16* __restrict__ kp,
    const _Float16* __restrict__ vpt, float* __restrict__ out)
{
  __shared__ __align__(128) _Float16 ks[2][64 * 64];   // [t][c] double-buffered
  __shared__ __align__(128) _Float16 vs[2][64 * 64];   // [c][t] double-buffered
  __shared__ __align__(32)  _Float16 ps[8 * 16 * 64];  // per-wave P scratch [m][t]

  const int tid  = threadIdx.x;
  const int lane = tid & 31;
  const int wid  = tid >> 5;
  const int l16  = lane & 15;
  const int hsel = (lane < 16) ? 0 : 1;
  const int ah    = blockIdx.y;                 // a*16 + h
  const int mbase = blockIdx.x * 128;
  const int mrow0 = mbase + wid * 16;

  const _Float16* kbase = kp  + (size_t)ah * 2048 * 64;
  const _Float16* vbase = vpt + (size_t)ah * 64 * 2048;

  // Kick off DMA of chunk 0 while we load Q fragments
  if (wid == 0) {
    tdm_load_f16((unsigned)(uintptr_t)&ks[0][0], kbase, 4096u, 0u, 4096ull, 0u, 4096ull);
    tdm_load_f16((unsigned)(uintptr_t)&vs[0][0], vbase, 64u, 64u, 2048ull, 64u, 2048ull);
  }

  // Resident Q fragments (16 rows x 64 c = two K=32 A-frags)
  v16h qf[2];
  {
    const _Float16* qrow = qp + ((size_t)ah * 2048 + mrow0 + l16) * 64;
    #pragma unroll
    for (int kk = 0; kk < 2; ++kk) {
      int o = kk * 32 + hsel * 16;
      *(v8h*)&qf[kk]       = *(const v8h*)(qrow + o);
      *((v8h*)&qf[kk] + 1) = *(const v8h*)(qrow + o + 8);
    }
  }

  v8f of[4] = {};
  float mrun[8], lrun[8];
  #pragma unroll
  for (int r = 0; r < 8; ++r) { mrun[r] = -__builtin_huge_valf(); lrun[r] = 0.0f; }

  _Float16* pw = ps + wid * 1024;

  for (int it = 0; it < 32; ++it) {
    const int p = it & 1;
    // Chunk `it` complete? (only chunk `it` is outstanding here; in-order)
    if (wid == 0) __builtin_amdgcn_s_wait_tensorcnt(0);
    __syncthreads();   // all waves see buffer p; prior compute on buffer 1-p done
    if (wid == 0 && it + 1 < 32) {
      int tn = (it + 1) * 64;
      tdm_load_f16((unsigned)(uintptr_t)&ks[1 - p][0], kbase + (size_t)tn * 64,
                   4096u, 0u, 4096ull, 0u, 4096ull);
      tdm_load_f16((unsigned)(uintptr_t)&vs[1 - p][0], vbase + tn,
                   64u, 64u, 2048ull, 64u, 2048ull);
    }

    // S = Q @ K^T   (16 x 64 scores per wave, unscaled — faithful to ref)
    v8f sf[4] = {};
    #pragma unroll
    for (int kk = 0; kk < 2; ++kk) {
      #pragma unroll
      for (int nt = 0; nt < 4; ++nt) {
        int trow = nt * 16 + l16;
        v16h bf = *(const v16h*)&ks[p][trow * 64 + kk * 32 + hsel * 16];
        sf[nt] = WMMA_F16(qf[kk], bf, sf[nt]);
      }
    }

    // Online softmax: row max (lanes sharing a row form a 16-lane group)
    float mx[8];
    #pragma unroll
    for (int r = 0; r < 8; ++r)
      mx[r] = fmaxf(fmaxf(sf[0][r], sf[1][r]), fmaxf(sf[2][r], sf[3][r]));
    #pragma unroll
    for (int d = 1; d < 16; d <<= 1)
      #pragma unroll
      for (int r = 0; r < 8; ++r)
        mx[r] = fmaxf(mx[r], __shfl_xor(mx[r], d, 32));

    float scale[8];
    #pragma unroll
    for (int r = 0; r < 8; ++r) {
      float mn = fmaxf(mrun[r], mx[r]);
      scale[r] = __expf(mrun[r] - mn);
      mrun[r] = mn;
    }

    // P = exp(S - m_new): accumulate row sums, spill to LDS for re-layout
    float rsum[8] = {0, 0, 0, 0, 0, 0, 0, 0};
    #pragma unroll
    for (int nt = 0; nt < 4; ++nt) {
      int tcol = nt * 16 + l16;
      #pragma unroll
      for (int r = 0; r < 8; ++r) {
        float pv = __expf(sf[nt][r] - mrun[r]);
        rsum[r] += pv;
        pw[(r + hsel * 8) * 64 + tcol] = (_Float16)pv;
      }
    }
    #pragma unroll
    for (int d = 1; d < 16; d <<= 1)
      #pragma unroll
      for (int r = 0; r < 8; ++r)
        rsum[r] += __shfl_xor(rsum[r], d, 32);
    #pragma unroll
    for (int r = 0; r < 8; ++r) lrun[r] = lrun[r] * scale[r] + rsum[r];

    // Rescale running O
    #pragma unroll
    for (int ct = 0; ct < 4; ++ct)
      #pragma unroll
      for (int r = 0; r < 8; ++r)
        of[ct][r] *= scale[r];

    // Reload P as A-fragments (per-wave scratch; LDS ops are in-order)
    asm volatile("s_wait_dscnt 0" ::: "memory");
    v16h pf[2];
    #pragma unroll
    for (int kk = 0; kk < 2; ++kk) {
      int o = kk * 32 + hsel * 16;
      *(v8h*)&pf[kk]       = *(const v8h*)&pw[l16 * 64 + o];
      *((v8h*)&pf[kk] + 1) = *(const v8h*)&pw[l16 * 64 + o + 8];
    }

    // O += P @ V
    #pragma unroll
    for (int ct = 0; ct < 4; ++ct)
      #pragma unroll
      for (int kk = 0; kk < 2; ++kk) {
        int crow = ct * 16 + l16;
        v16h bf = *(const v16h*)&vs[p][crow * 64 + kk * 32 + hsel * 16];
        of[ct] = WMMA_F16(pf[kk], bf, of[ct]);
      }
    // No trailing barrier: next iteration's wait+barrier protects both buffers.
  }

  // Normalize and write fp32 output [a][m][h*64+c]
  const int a = ah >> 4, h = ah & 15;
  #pragma unroll
  for (int r = 0; r < 8; ++r) {
    float inv = 1.0f / lrun[r];
    int m = mrow0 + r + hsel * 8;
    float* orow = out + ((size_t)(a * 2048 + m)) * 1024 + h * 64;
    #pragma unroll
    for (int ct = 0; ct < 4; ++ct)
      orow[ct * 16 + l16] = of[ct][r] * inv;
  }
}

// ---------------------------------------------------------------------------
extern "C" void kernel_launch(void* const* d_in, const int* in_sizes, int n_in,
                              void* d_out, int out_size, void* d_ws, size_t ws_size,
                              hipStream_t stream) {
  const float* q  = (const float*)d_in[0];
  const float* k  = (const float*)d_in[1];
  const float* v  = (const float*)d_in[2];
  const float* Wq = (const float*)d_in[3];
  const float* bq = (const float*)d_in[4];
  const float* Wk = (const float*)d_in[5];
  const float* bk = (const float*)d_in[6];
  const float* Wv = (const float*)d_in[7];
  const float* bv = (const float*)d_in[8];
  float* out = (float*)d_out;

  // Workspace: QP, KP (head-major [ah][seq][64]) and V^T ([ah][64][seq]),
  // f16: 3 x 16 MB = 48 MB.
  const size_t PROJ_ELEMS = (size_t)4 * 16 * 2048 * 64;
  _Float16* qp  = (_Float16*)d_ws;
  _Float16* kp  = qp + PROJ_ELEMS;
  _Float16* vpt = kp + PROJ_ELEMS;

  // Phase 1: three projection GEMMs (mode = blockIdx.z)
  proj_kernel<<<dim3(1024 / 128, 8192 / 64, 3), 256, 0, stream>>>(
      q, k, v, Wq, bq, Wk, bk, Wv, bv, qp, kp, vpt);

  // Phase 2: flash attention, one block per (128 Q-rows, head)
  attn_kernel<<<dim3(2048 / 128, 4 * 16), 256, 0, stream>>>(qp, kp, vpt, out);
}
